// TransformerBlock_11227044512446
// MI455X (gfx1250) — matrix operand order repaired and checked by reading, not compiled, not run on hardware
//
#include <hip/hip_runtime.h>

// ---------------------------------------------------------------------------
// Transformer block for MI455X (gfx1250, wave32, WMMA).
// bf16 inputs + f32 WMMA accumulation; flash-style fused attention.
// Round 2: software-pipelined GEMM K-loop + global prefetch; flash Q tile
// staged with gfx1250 async global->LDS (ASYNCcnt) instead of VGPR bounce.
// ---------------------------------------------------------------------------

typedef unsigned int   u32;
typedef unsigned short u16;

typedef __attribute__((ext_vector_type(16))) __bf16 bf16x16;
typedef __attribute__((ext_vector_type(8)))  float  f32x8;
typedef __attribute__((ext_vector_type(4)))  u32    u32x4;

#define T_SEQ  2048
#define EMBD   256
#define NHEADS 8
#define HE_    (NHEADS * EMBD)   // 2048
#define FFD    (4 * EMBD)        // 1024

union FragCast { bf16x16 v; u32x4 u[2]; };

// float -> bf16, round-to-nearest-even
__device__ __forceinline__ u16 f2bf(float f) {
  u32 u = __float_as_uint(f);
  u32 r = u + 0x7FFFu + ((u >> 16) & 1u);
  return (u16)(r >> 16);
}

// D = A*B + C, 16x16x32 bf16 -> f32
__device__ __forceinline__ f32x8 wmma_bf16(bf16x16 a, bf16x16 b, f32x8 c) {
  return __builtin_amdgcn_wmma_f32_16x16x32_bf16(
      /*neg_a=*/false, a, /*neg_b=*/false, b,
      /*c_mod=*/(short)0, c, /*reuse_a=*/false, /*reuse_b=*/false);
}

// A fragment: 16x32 bf16 tile (MxK), A row-major with leading dim ld.
// lanes 0-15: row M=lane,    K = {k0..k0+7} , {k0+16..k0+23}
// lanes16-31: row M=lane-16, K = {k0+8..+15}, {k0+24..+31}
__device__ __forceinline__ bf16x16 load_a_frag(const u16* base, int ld,
                                               int m0, int k0, int lane) {
  int r  = lane & 15;
  int kk = (lane >> 4) << 3;                  // 0 or 8
  const u16* p = base + (size_t)(m0 + r) * ld + k0 + kk;
  FragCast f;
  f.u[0] = *(const u32x4*)p;
  f.u[1] = *(const u32x4*)(p + 16);
  return f.v;
}

// B fragment: 32x16 bf16 tile (KxN) sourced from [N,K] row-major (ld = K stride).
// lanes 0-15: col N=lane,    K = {k0..k0+7} , {k0+16..k0+23}
// lanes16-31: col N=lane-16, K = {k0+8..+15}, {k0+24..+31}
__device__ __forceinline__ bf16x16 load_b_frag(const u16* base, int ld,
                                               int n0, int k0, int lane) {
  int n  = lane & 15;
  int kk = (lane >> 4) << 3;                  // 0 or 8
  const u16* p = base + (size_t)(n0 + n) * ld + k0 + kk;
  FragCast f;
  f.u[0] = *(const u32x4*)p;
  f.u[1] = *(const u32x4*)(p + 16);
  return f.v;
}

// ---------------------------------------------------------------------------
// f32 -> bf16 conversion
// ---------------------------------------------------------------------------
__global__ void cvt_bf16_kernel(const float* __restrict__ in,
                                u16* __restrict__ out, int n) {
  for (int i = blockIdx.x * blockDim.x + threadIdx.x; i < n;
       i += gridDim.x * blockDim.x)
    out[i] = f2bf(in[i]);
}

// v[b][t][h*E+e] -> vT[(b*H+h)*E+e][t]   (t contiguous for PV B-fragments)
__global__ void transpose_v_kernel(const u16* __restrict__ vb,
                                   u16* __restrict__ vtb) {
  const int total = 2 * T_SEQ * HE_;
  for (int i = blockIdx.x * blockDim.x + threadIdx.x; i < total;
       i += gridDim.x * blockDim.x) {
    int e = i & (EMBD - 1);
    int h = (i / EMBD) & (NHEADS - 1);
    int t = (i / HE_) & (T_SEQ - 1);
    int b = i / (T_SEQ * HE_);
    vtb[((size_t)(b * NHEADS + h) * EMBD + e) * T_SEQ + t] = (u16)vb[i];
  }
}

// ---------------------------------------------------------------------------
// C[M,N] = act(scale * A[M,K] @ B[N,K]^T + bias)   (torch Linear layout)
// block = 128 threads (4 waves); each wave computes a 32x32 tile (2x2 WMMA).
// Software-pipelined K loop: fragments for step k+1 are issued before the
// WMMAs of step k so VMEM latency overlaps the matrix pipe; explicit
// global_prefetch for step k+2 keeps L2->L0 traffic ahead of the math.
// OUTMODE: 0 = f32, 1 = bf16
// ---------------------------------------------------------------------------
template <int OUTMODE, bool RELU, bool HASBIAS>
__global__ __launch_bounds__(128) void gemm_bt_kernel(
    const u16* __restrict__ A, const u16* __restrict__ Bw,
    void* __restrict__ Cout, const float* __restrict__ bias, float scale,
    int K, int lda, int ldb, int ldc) {
  const int lane = threadIdx.x & 31;
  const int wave = threadIdx.x >> 5;
  const int m0 = blockIdx.x * 64 + (wave & 1) * 32;
  const int n0 = blockIdx.y * 64 + (wave >> 1) * 32;

  f32x8 acc00 = {}, acc01 = {}, acc10 = {}, acc11 = {};

  bf16x16 a0 = load_a_frag(A, lda, m0,      0, lane);
  bf16x16 a1 = load_a_frag(A, lda, m0 + 16, 0, lane);
  bf16x16 b0 = load_b_frag(Bw, ldb, n0,      0, lane);
  bf16x16 b1 = load_b_frag(Bw, ldb, n0 + 16, 0, lane);

  for (int k0 = 32; k0 < K; k0 += 32) {
    // issue next-step fragment loads first (latency hiding)
    bf16x16 na0 = load_a_frag(A, lda, m0,      k0, lane);
    bf16x16 na1 = load_a_frag(A, lda, m0 + 16, k0, lane);
    bf16x16 nb0 = load_b_frag(Bw, ldb, n0,      k0, lane);
    bf16x16 nb1 = load_b_frag(Bw, ldb, n0 + 16, k0, lane);
    if (k0 + 32 < K) {
      __builtin_prefetch(A  + (size_t)(m0 + (lane & 15)) * lda + k0 + 32, 0, 1);
      __builtin_prefetch(Bw + (size_t)(n0 + (lane & 15)) * ldb + k0 + 32, 0, 1);
    }
    acc00 = wmma_bf16(a0, b0, acc00);
    acc01 = wmma_bf16(a0, b1, acc01);
    acc10 = wmma_bf16(a1, b0, acc10);
    acc11 = wmma_bf16(a1, b1, acc11);
    a0 = na0; a1 = na1; b0 = nb0; b1 = nb1;
  }
  acc00 = wmma_bf16(a0, b0, acc00);
  acc01 = wmma_bf16(a0, b1, acc01);
  acc10 = wmma_bf16(a1, b0, acc10);
  acc11 = wmma_bf16(a1, b1, acc11);

  const int nlo   = lane & 15;
  const int rbase = (lane >> 4) * 8;          // C layout: lanes>=16 hold M=8..15
  f32x8 accs[4] = {acc00, acc01, acc10, acc11};
#pragma unroll
  for (int t = 0; t < 4; ++t) {
    const int ti = t >> 1, tj = t & 1;
    const int col = n0 + tj * 16 + nlo;
    const float bv = HASBIAS ? bias[col] : 0.0f;
#pragma unroll
    for (int r = 0; r < 8; ++r) {
      float v = accs[t][r] * scale + bv;
      if (RELU) v = fmaxf(v, 0.0f);
      const int row = m0 + ti * 16 + rbase + r;
      if (OUTMODE == 0)
        ((float*)Cout)[(size_t)row * ldc + col] = v;
      else
        ((u16*)Cout)[(size_t)row * ldc + col] = f2bf(v);
    }
  }
}

// ---------------------------------------------------------------------------
// Flash attention: one wave32 per (b,h, 16-row q-tile).
// Q tile staged in LDS via gfx1250 async global->LDS DMA (no VGPR bounce,
// tracked by ASYNCcnt); K,V streamed from L2 (entire bf16 QKV set fits 192MB).
// Online softmax (f32) with cross-lane reductions over 16-lane half-groups
// (matches the C/D fragment layout). Probabilities bounced through a 1KB LDS
// tile to reshape C-layout -> A-fragment layout for the PV WMMA.
// ---------------------------------------------------------------------------
__global__ __launch_bounds__(32) void flash_attn_kernel(
    const u16* __restrict__ q, const u16* __restrict__ k,
    const u16* __restrict__ vt, u16* __restrict__ out) {
  const int lane = threadIdx.x;
  const int qt = blockIdx.x;          // q tile: 0..T/16-1
  const int bh = blockIdx.y;          // 0..B*H-1
  const int b  = bh >> 3;
  const int h  = bh & 7;
  const int m0 = qt * 16;

  const u16* qptr = q  + (size_t)b * T_SEQ * HE_ + (size_t)h * EMBD;
  const u16* kptr = k  + (size_t)b * T_SEQ * HE_ + (size_t)h * EMBD;
  const u16* vptr = vt + (size_t)bh * EMBD * T_SEQ;
  u16*       optr = out + (size_t)b * T_SEQ * HE_ + (size_t)h * EMBD;

  __shared__ __align__(16) u16 sQ[16 * EMBD];   // 8 KB
  __shared__ __align__(16) u16 sP[16 * 32];     // 1 KB

  // Stage the 16x256 Q tile straight into LDS with async DMA loads:
  // GLOBAL_LOAD_ASYNC_TO_LDS_B128 moves 16B per lane memory->LDS without
  // touching VGPRs; completion is tracked by ASYNCcnt.
#pragma unroll
  for (int i = 0; i < 16; ++i) {
    int idx = i * 32 + lane;                    // 512 chunks of 8 halfwords
    int row = idx >> 5;
    int c8  = idx & 31;
    const u16* gsrc = qptr + (size_t)(m0 + row) * HE_ + c8 * 8;
    u32 ldsoff = (u32)(unsigned long long)(const void*)(sQ + row * EMBD + c8 * 8);
    asm volatile("global_load_async_to_lds_b128 %0, %1, off"
                 :
                 : "v"(ldsoff), "v"(gsrc)
                 : "memory");
  }
  asm volatile("s_wait_asynccnt 0x0" ::: "memory");
  __syncthreads();

  f32x8 vzero = {};
  float mrow[8], lrow[8];
  f32x8 o[16];
#pragma unroll
  for (int r = 0; r < 8; ++r) { mrow[r] = -3.0e38f; lrow[r] = 0.0f; }
#pragma unroll
  for (int n = 0; n < 16; ++n) o[n] = vzero;

  const int rbase = (lane >> 4) * 8;
  const int nlo   = lane & 15;

  for (int kt = 0; kt < T_SEQ / 32; ++kt) {
    const int kr0 = kt * 32;
    // S = Q @ K^T for 32 key rows (two 16-wide subtiles), K-dim = 256
    f32x8 s0 = vzero, s1 = vzero;
#pragma unroll
    for (int c = 0; c < 8; ++c) {
      bf16x16 qf  = load_a_frag(sQ, EMBD, 0, c * 32, lane);
      bf16x16 kf0 = load_b_frag(kptr, HE_, kr0,      c * 32, lane);
      bf16x16 kf1 = load_b_frag(kptr, HE_, kr0 + 16, c * 32, lane);
      s0 = wmma_bf16(qf, kf0, s0);
      s1 = wmma_bf16(qf, kf1, s1);
    }
    // row-max over the 32 new columns (reduce across 16-lane half-groups)
    float rm[8];
#pragma unroll
    for (int r = 0; r < 8; ++r) rm[r] = fmaxf(s0[r], s1[r]);
#pragma unroll
    for (int off = 8; off >= 1; off >>= 1)
#pragma unroll
      for (int r = 0; r < 8; ++r)
        rm[r] = fmaxf(rm[r], __shfl_xor(rm[r], off, 32));

    float corr[8], rs[8];
    f32x8 p0, p1;
#pragma unroll
    for (int r = 0; r < 8; ++r) {
      float nm = fmaxf(mrow[r], rm[r]);
      corr[r] = __expf(mrow[r] - nm);
      p0[r]   = __expf(s0[r] - nm);
      p1[r]   = __expf(s1[r] - nm);
      rs[r]   = p0[r] + p1[r];
      mrow[r] = nm;
    }
#pragma unroll
    for (int off = 8; off >= 1; off >>= 1)
#pragma unroll
      for (int r = 0; r < 8; ++r) rs[r] += __shfl_xor(rs[r], off, 32);
#pragma unroll
    for (int r = 0; r < 8; ++r) lrow[r] = lrow[r] * corr[r] + rs[r];

    // stash P (bf16, row-major 16x32) in LDS, re-read as A-fragment
#pragma unroll
    for (int r = 0; r < 8; ++r) {
      sP[(rbase + r) * 32 + nlo]      = f2bf(p0[r]);
      sP[(rbase + r) * 32 + 16 + nlo] = f2bf(p1[r]);
    }
    __syncthreads();
    bf16x16 pf = load_a_frag(sP, 32, 0, 0, lane);

    // O = O*corr + P @ V_chunk   (V^T rows are e, t contiguous)
#pragma unroll
    for (int n = 0; n < 16; ++n) {
      f32x8 acc = o[n];
#pragma unroll
      for (int r = 0; r < 8; ++r) acc[r] *= corr[r];
      bf16x16 vf = load_b_frag(vptr, T_SEQ, n * 16, kr0, lane);
      o[n] = wmma_bf16(pf, vf, acc);
    }
    __syncthreads();
  }

  float inv[8];
#pragma unroll
  for (int r = 0; r < 8; ++r) inv[r] = 1.0f / lrow[r];
#pragma unroll
  for (int n = 0; n < 16; ++n)
#pragma unroll
    for (int r = 0; r < 8; ++r) {
      const int row = m0 + rbase + r;
      optr[(size_t)row * HE_ + n * 16 + nlo] = f2bf(o[n][r] * inv[r]);
    }
}

// ---------------------------------------------------------------------------
// y = LayerNorm(res + xin) * g + be ; one wave per 256-element row.
// Writes f32 always; bf16 copy optionally (for the following bf16 GEMM).
// ---------------------------------------------------------------------------
__global__ __launch_bounds__(32) void add_ln_kernel(
    const float* __restrict__ res, const float* __restrict__ xin,
    const float* __restrict__ g, const float* __restrict__ be,
    float* __restrict__ yf, u16* __restrict__ yb) {
  const int row  = blockIdx.x;
  const int lane = threadIdx.x;
  const float* a = res + (size_t)row * EMBD;
  const float* x = xin + (size_t)row * EMBD;

  float v[8];
  float s = 0.0f;
#pragma unroll
  for (int i = 0; i < 8; ++i) {
    v[i] = a[lane + 32 * i] + x[lane + 32 * i];
    s += v[i];
  }
#pragma unroll
  for (int off = 16; off >= 1; off >>= 1) s += __shfl_xor(s, off, 32);
  const float mu = s * (1.0f / EMBD);

  float var = 0.0f;
#pragma unroll
  for (int i = 0; i < 8; ++i) { float d = v[i] - mu; var += d * d; }
#pragma unroll
  for (int off = 16; off >= 1; off >>= 1) var += __shfl_xor(var, off, 32);
  const float rstd = rsqrtf(var * (1.0f / EMBD) + 1e-5f);

#pragma unroll
  for (int i = 0; i < 8; ++i) {
    const int c = lane + 32 * i;
    const float y = (v[i] - mu) * rstd * g[c] + be[c];
    yf[(size_t)row * EMBD + c] = y;
    if (yb) yb[(size_t)row * EMBD + c] = f2bf(y);
  }
}

// ---------------------------------------------------------------------------
extern "C" void kernel_launch(void* const* d_in, const int* in_sizes, int n_in,
                              void* d_out, int out_size, void* d_ws,
                              size_t ws_size, hipStream_t stream) {
  (void)in_sizes; (void)n_in; (void)out_size; (void)ws_size;

  const float* x   = (const float*)d_in[0];
  const float* wk  = (const float*)d_in[1];
  const float* wq  = (const float*)d_in[2];
  const float* wv  = (const float*)d_in[3];
  const float* wu  = (const float*)d_in[4];
  const float* bu  = (const float*)d_in[5];
  const float* g1  = (const float*)d_in[6];
  const float* be1 = (const float*)d_in[7];
  const float* g2  = (const float*)d_in[8];
  const float* be2 = (const float*)d_in[9];
  const float* w1  = (const float*)d_in[10];
  const float* b1  = (const float*)d_in[11];
  const float* w2  = (const float*)d_in[12];
  const float* b2  = (const float*)d_in[13];

  const int M = 2 * T_SEQ;                        // 4096 rows (B*T)

  // carve scratch (~120 MB needed), 256B aligned
  char* p = (char*)d_ws;
  auto carve = [&](size_t bytes) {
    char* r = p;
    p += (bytes + 255) & ~(size_t)255;
    return r;
  };
  u16*   xb  = (u16*)carve((size_t)M * EMBD * 2);
  u16*   wqb = (u16*)carve((size_t)HE_ * EMBD * 2);
  u16*   wkb = (u16*)carve((size_t)HE_ * EMBD * 2);
  u16*   wvb = (u16*)carve((size_t)HE_ * EMBD * 2);
  u16*   wub = (u16*)carve((size_t)EMBD * HE_ * 2);
  u16*   w1b = (u16*)carve((size_t)FFD * EMBD * 2);
  u16*   w2b = (u16*)carve((size_t)EMBD * FFD * 2);
  u16*   qb  = (u16*)carve((size_t)M * HE_ * 2);
  u16*   kb  = (u16*)carve((size_t)M * HE_ * 2);
  u16*   vb  = (u16*)carve((size_t)M * HE_ * 2);
  u16*   vtb = (u16*)carve((size_t)M * HE_ * 2);
  u16*   ab  = (u16*)carve((size_t)M * HE_ * 2);      // attention out (bf16)
  float* att = (float*)carve((size_t)M * EMBD * 4);   // attended (f32)
  float* x1f = (float*)carve((size_t)M * EMBD * 4);
  u16*   x1b = (u16*)carve((size_t)M * EMBD * 2);
  u16*   f1b = (u16*)carve((size_t)M * FFD * 2);
  float* f2  = (float*)carve((size_t)M * EMBD * 4);

  // bf16 copies of activations and weights
  cvt_bf16_kernel<<<1024, 256, 0, stream>>>(x,  xb,  M * EMBD);
  cvt_bf16_kernel<<<1024, 256, 0, stream>>>(wq, wqb, HE_ * EMBD);
  cvt_bf16_kernel<<<1024, 256, 0, stream>>>(wk, wkb, HE_ * EMBD);
  cvt_bf16_kernel<<<1024, 256, 0, stream>>>(wv, wvb, HE_ * EMBD);
  cvt_bf16_kernel<<<1024, 256, 0, stream>>>(wu, wub, EMBD * HE_);
  cvt_bf16_kernel<<<1024, 256, 0, stream>>>(w1, w1b, FFD * EMBD);
  cvt_bf16_kernel<<<1024, 256, 0, stream>>>(w2, w2b, EMBD * FFD);

  // QKV projections; q,k pre-scaled by EMB^-1/4 = 0.25
  dim3 gqkv(M / 64, HE_ / 64);
  gemm_bt_kernel<1, false, false><<<gqkv, 128, 0, stream>>>(
      xb, wqb, qb, nullptr, 0.25f, EMBD, EMBD, EMBD, HE_);
  gemm_bt_kernel<1, false, false><<<gqkv, 128, 0, stream>>>(
      xb, wkb, kb, nullptr, 0.25f, EMBD, EMBD, EMBD, HE_);
  gemm_bt_kernel<1, false, false><<<gqkv, 128, 0, stream>>>(
      xb, wvb, vb, nullptr, 1.0f, EMBD, EMBD, EMBD, HE_);

  // V -> V^T (per head, t contiguous) for the PV B-fragments
  transpose_v_kernel<<<4096, 256, 0, stream>>>(vb, vtb);

  // fused attention: softmax(QK^T) @ V, output bf16 [B,T,H*E]
  flash_attn_kernel<<<dim3(T_SEQ / 16, 2 * NHEADS), 32, 0, stream>>>(
      qb, kb, vtb, ab);

  // output projection + bias -> attended (f32)
  gemm_bt_kernel<0, false, true><<<dim3(M / 64, EMBD / 64), 128, 0, stream>>>(
      ab, wub, att, bu, 1.0f, HE_, HE_, HE_, EMBD);

  // x1 = LN(attended + x)
  add_ln_kernel<<<M, 32, 0, stream>>>(att, x, g1, be1, x1f, x1b);

  // ff = relu(x1 @ w1^T + b1) @ w2^T + b2
  gemm_bt_kernel<1, true, true><<<dim3(M / 64, FFD / 64), 128, 0, stream>>>(
      x1b, w1b, f1b, b1, 1.0f, EMBD, EMBD, EMBD, FFD);
  gemm_bt_kernel<0, false, true><<<dim3(M / 64, EMBD / 64), 128, 0, stream>>>(
      f1b, w2b, f2, b2, 1.0f, FFD, FFD, FFD, EMBD);

  // out = LN(ff + x1)
  add_ln_kernel<<<M, 32, 0, stream>>>(f2, x1f, g2, be2, (float*)d_out, nullptr);
}
